// GFLSTM_76424648065777
// MI455X (gfx1250) — compile-verified, matrix operand order checked by reading, not run on hardware
//
#include <hip/hip_runtime.h>
#include <hip/hip_bf16.h>
#include <stdint.h>

typedef unsigned short u16;
typedef __attribute__((ext_vector_type(16))) __bf16 v16bf;
typedef __attribute__((ext_vector_type(8)))  __bf16 v8bf;
typedef __attribute__((ext_vector_type(8)))  float  v8f;

#define BB   4096
#define DD   256
#define HH   256
#define SS   3
#define REPT 10
#define OUTN 44
#define CATW (SS*HH)          // 768
#define FEATW ((REPT-1)*HH)   // 2304
#define G4   (4*HH)           // 1024 packed gate columns (f,i,o,c)
#define G3   (3*HH)           // 768 packed hidden-gate columns (f,i,o)
#define PADK 40               // 32 payload + 8 pad halfwords: conflict-free rows

union FragU { v16bf v; v8bf h[2]; };

static __device__ __forceinline__ u16 f2bf(float f) {
  unsigned u; __builtin_memcpy(&u, &f, 4);
  u += 0x7FFFu + ((u >> 16) & 1u);          // round-to-nearest-even
  return (u16)(u >> 16);
}
static __device__ __forceinline__ float bf2f(u16 v) {
  unsigned u = ((unsigned)v) << 16;
  float f; __builtin_memcpy(&f, &u, 4);
  return f;
}
static __device__ __forceinline__ float sigm(float x) {
  return 1.0f / (1.0f + __expf(-x));
}

// ---------------------------------------------------------------------------
// bf16 WMMA GEMM:  C[M,N] (f32) = A[M,lda] * W[N,ldw]^T  (+ C if accum)
// Block = 256 threads (8 waves), tile 128(M) x 64(N). The 64x32 bf16 weight
// tile is double-buffered in LDS via async global->LDS copies (ASYNCcnt) and
// consumed by all 8 waves with ds_load_b128. A fragments are issued to global
// *before* the async-wait + barrier so their latency hides under LDS reads.
// K mult of 32, M mult of 128; N edge guarded (clamped rows, guarded stores).
// ---------------------------------------------------------------------------
__global__ __launch_bounds__(256)
void gemm_bf16(const u16* __restrict__ A, int lda,
               const u16* __restrict__ W, int ldw, int N,
               float* __restrict__ C, int ldc, int K, int accum)
{
  __shared__ u16 Bs[2][64 * PADK];

  const int tid   = threadIdx.x;
  const int lane  = tid & 31;
  const int wave  = tid >> 5;
  const int m0    = blockIdx.y * 128 + wave * 16;
  const int nb    = blockIdx.x * 64;
  const int frow  = lane & 15;          // fragment row (M for A, N for W)
  const int khalf = (lane >> 4) * 8;    // K sub-offset per ISA layout
  const int mhalf = (lane >> 4) * 8;    // C/D: lanes 16-31 hold M+8
  const int ncol  = lane & 15;          // C/D column within tile

  // Staging geometry: 256 threads move a 64-row x 32-halfword tile,
  // one b128 (8 halfwords) per thread per K-step.
  const int srow = tid >> 2;            // 0..63
  const int sseg = (tid & 3) * 8;       // 0,8,16,24
  int swr = nb + srow; if (swr >= N) swr = N - 1;   // clamp edge (N=44 head)
  const u16* sgp = W + (size_t)swr * ldw + sseg;

  auto stage = [&](int kt, int buf) {
    unsigned ldsoff = (unsigned)(uintptr_t)(&Bs[buf][srow * PADK + sseg]);
    const u16* g = sgp + kt;
    asm volatile("global_load_async_to_lds_b128 %0, %1, off"
                 :: "v"(ldsoff), "v"(g) : "memory");
  };

  const v8f vzero = {0.f,0.f,0.f,0.f,0.f,0.f,0.f,0.f};
  v8f acc[4];
#pragma unroll
  for (int j = 0; j < 4; ++j) acc[j] = vzero;

  if (accum) {
#pragma unroll
    for (int j = 0; j < 4; ++j) {
      int col = nb + j * 16 + ncol;
      if (col < N) {
#pragma unroll
        for (int r = 0; r < 8; ++r)
          acc[j][r] = C[(size_t)(m0 + mhalf + r) * ldc + col];
      }
    }
  }

  const u16* Arow = A + (size_t)(m0 + frow) * lda;

  stage(0, 0);                               // prologue: tile 0 in flight
  for (int k = 0, it = 0; k < K; k += 32, ++it) {
    const int  buf  = it & 1;
    const bool more = (k + 32) < K;

    FragU a;                                 // A frag: issue early (global)
    a.h[0] = *(const v8bf*)(const void*)(Arow + k + khalf);
    a.h[1] = *(const v8bf*)(const void*)(Arow + k + 16 + khalf);

    if (more) stage(k + 32, buf ^ 1);        // prefetch next tile (async)
    if (more) asm volatile("s_wait_asynccnt 0x1" ::: "memory");  // cur done
    else      asm volatile("s_wait_asynccnt 0x0" ::: "memory");
    __syncthreads();                         // cur tile visible to all waves

#pragma unroll
    for (int j = 0; j < 4; ++j) {            // B fragments: LDS, shared 8x
      const u16* brow = &Bs[buf][(j * 16 + frow) * PADK];
      FragU b;
      b.h[0] = *(const v8bf*)(const void*)(brow + khalf);
      b.h[1] = *(const v8bf*)(const void*)(brow + 16 + khalf);
      acc[j] = __builtin_amdgcn_wmma_f32_16x16x32_bf16(
          false, a.v, false, b.v, (short)0, acc[j], false, false);
    }
    __syncthreads();                         // reads done before overwrite
  }

#pragma unroll
  for (int j = 0; j < 4; ++j) {
    int col = nb + j * 16 + ncol;
    if (col < N) {
#pragma unroll
      for (int r = 0; r < 8; ++r)
        C[(size_t)(m0 + mhalf + r) * ldc + col] = acc[j][r];
    }
  }
}

// ---------------------------------------------------------------------------
__global__ void f32_to_bf16_k(const float* __restrict__ src,
                              u16* __restrict__ dst, int n) {
  int i = blockIdx.x * blockDim.x + threadIdx.x;
  if (i < n) dst[i] = f2bf(src[i]);
}

// Pack [Wi_f;Wi_i;Wi_o;Wi_c] -> dst[S][1024][256] (bf16), per-layer blocks.
__global__ void pack4_k(const float* __restrict__ Wf, const float* __restrict__ Wi,
                        const float* __restrict__ Wo, const float* __restrict__ Wc,
                        u16* __restrict__ dst) {
  int i = blockIdx.x * blockDim.x + threadIdx.x;
  if (i >= SS * G4 * DD) return;
  int l = i / (G4 * DD);
  int r = (i / DD) % G4;
  int k = i % DD;
  int g = r >> 8, rr = r & 255;
  size_t s = ((size_t)l * HH + rr) * DD + k;
  float v = (g == 0) ? Wf[s] : (g == 1) ? Wi[s] : (g == 2) ? Wo[s] : Wc[s];
  dst[i] = f2bf(v);
}

// Pack [Wh_f;Wh_i;Wh_o] -> dst[S][768][256] (bf16), per-layer blocks.
__global__ void pack3_k(const float* __restrict__ Wf, const float* __restrict__ Wi,
                        const float* __restrict__ Wo, u16* __restrict__ dst) {
  int i = blockIdx.x * blockDim.x + threadIdx.x;
  if (i >= SS * G3 * HH) return;
  int l = i / (G3 * HH);
  int r = (i / HH) % G3;
  int k = i % HH;
  int g = r >> 8, rr = r & 255;
  size_t s = ((size_t)l * HH + rr) * HH + k;
  float v = (g == 0) ? Wf[s] : (g == 1) ? Wi[s] : Wo[s];
  dst[i] = f2bf(v);
}

// hcat[b, l*H+h] = hbuf[l, b, h]   (layer-major concat)
__global__ void build_hcat_k(const u16* __restrict__ hbuf,
                             u16* __restrict__ hcat) {
  int i = blockIdx.x * blockDim.x + threadIdx.x;
  if (i >= SS * BB * HH) return;
  int l = i / (BB * HH);
  int rem = i - l * (BB * HH);
  int b = rem / HH;
  int h = rem - b * HH;
  hcat[(size_t)b * CATW + l * HH + h] = hbuf[i];
}

// g[b,s] = sigmoid(x·Wi_g[s] + bi_g[s] + hcat·Wh_g[s] + bh_g[s]); tiny FLOPs
__global__ void g_gate_k(const u16* __restrict__ xin, const u16* __restrict__ hcat,
                         const u16* __restrict__ Wig, const float* __restrict__ big,
                         const u16* __restrict__ Whg, const float* __restrict__ bhg,
                         float* __restrict__ g) {
  int i = blockIdx.x * blockDim.x + threadIdx.x;
  if (i >= BB * SS) return;
  int b = i / SS, s = i - b * SS;
  float acc = big[s] + bhg[s];
  const u16* xr = xin + (size_t)b * DD;
  const u16* wr = Wig + (size_t)s * DD;
  for (int k = 0; k < DD; ++k) acc += bf2f(xr[k]) * bf2f(wr[k]);
  const u16* hr = hcat + (size_t)b * CATW;
  const u16* wh = Whg + (size_t)s * CATW;
  for (int k = 0; k < CATW; ++k) acc += bf2f(hr[k]) * bf2f(wh[k]);
  g[i] = sigm(acc);
}

// Fused gate nonlinearities + gated cross-layer sum + cell/hidden update.
// P4 holds packed pre-activations [B,1024]: cols [f | i | o | c-input].
__global__ void combine_k(const float* __restrict__ P4,
                          const float* __restrict__ aux3, const float* __restrict__ g,
                          const float* __restrict__ bif, const float* __restrict__ bhf,
                          const float* __restrict__ bii, const float* __restrict__ bhi,
                          const float* __restrict__ bio, const float* __restrict__ bho,
                          const float* __restrict__ bic, const float* __restrict__ bhc,
                          float* __restrict__ c, u16* __restrict__ hout,
                          u16* __restrict__ feat /* may be null; ld=FEATW */) {
  int i = blockIdx.x * blockDim.x + threadIdx.x;
  if (i >= BB * HH) return;
  int b = i / HH, h = i - b * HH;
  const float* pr = P4 + (size_t)b * G4;
  float fg = sigm(pr[h]            + bif[h] + bhf[h]);
  float ig = sigm(pr[HH + h]       + bii[h] + bhi[h]);
  float og = sigm(pr[2 * HH + h]   + bio[h] + bho[h]);
  const float* ar = aux3 + (size_t)b * CATW;
  const float* gr = g + (size_t)b * SS;
  float aux = 0.f;
#pragma unroll
  for (int s = 0; s < SS; ++s) aux += gr[s] * (ar[s * HH + h] + bhc[s * HH + h]);
  float ct = tanhf(pr[3 * HH + h] + bic[h] + aux);
  float cv = fg * c[i] + ig * ct;
  c[i] = cv;
  float hv = og * cv;
  hout[i] = f2bf(hv);
  if (feat) feat[(size_t)b * FEATW + h] = f2bf(hv);
}

__global__ void final_out_k(const float* __restrict__ P,
                            const float* __restrict__ blast,
                            float* __restrict__ out) {
  int i = blockIdx.x * blockDim.x + threadIdx.x;
  if (i >= BB * OUTN) return;
  int o = i % OUTN;
  out[i] = sigm(P[i] + blast[o]);
}

// ---------------------------------------------------------------------------
extern "C" void kernel_launch(void* const* d_in, const int* in_sizes, int n_in,
                              void* d_out, int out_size, void* d_ws, size_t ws_size,
                              hipStream_t stream) {
  (void)in_sizes; (void)n_in; (void)out_size; (void)ws_size;

  size_t off = 0;
  auto walloc = [&](size_t bytes) -> void* {
    off = (off + 255) & ~(size_t)255;
    void* p = (char*)d_ws + off;
    off += bytes;
    return p;
  };

  // Activations / state
  u16*   xb    = (u16*)  walloc((size_t)BB * DD * 2);
  u16*   hbuf  = (u16*)  walloc((size_t)SS * BB * HH * 2);
  float* cbuf  = (float*)walloc((size_t)SS * BB * HH * 4);
  u16*   hcat  = (u16*)  walloc((size_t)BB * CATW * 2);
  float* P4    = (float*)walloc((size_t)BB * G4 * 4);
  float* aux3  = (float*)walloc((size_t)BB * CATW * 4);
  float* gbuf  = (float*)walloc((size_t)BB * SS * 4);
  u16*   feat  = (u16*)  walloc((size_t)BB * FEATW * 2);
  float* lastP = (float*)walloc((size_t)BB * OUTN * 4);
  // bf16 weights (gate weights packed: fewer, larger GEMMs)
  u16* wPin  = (u16*)walloc((size_t)SS * G4 * DD * 2);   // [Wi_f;Wi_i;Wi_o;Wi_c]
  u16* wPhid = (u16*)walloc((size_t)SS * G3 * HH * 2);   // [Wh_f;Wh_i;Wh_o]
  u16* wWig  = (u16*)walloc((size_t)SS * SS * DD * 2);
  u16* wWhg  = (u16*)walloc((size_t)SS * SS * CATW * 2);
  u16* wWhc  = (u16*)walloc((size_t)SS * CATW * CATW * 2);
  u16* wWlst = (u16*)walloc((size_t)OUTN * FEATW * 2);

  auto cvt = [&](const void* src, u16* dst, int n) {
    f32_to_bf16_k<<<(n + 255) / 256, 256, 0, stream>>>((const float*)src, dst, n);
  };

  // One-time per-call conversions / packing (cheap vs ~260 GFLOP of GEMMs)
  cvt(d_in[0],  xb,   BB * DD);
  cvt(d_in[1],  hbuf, SS * BB * HH);
  hipMemcpyAsync(cbuf, d_in[2], (size_t)SS * BB * HH * 4,
                 hipMemcpyDeviceToDevice, stream);
  pack4_k<<<(SS * G4 * DD + 255) / 256, 256, 0, stream>>>(
      (const float*)d_in[3], (const float*)d_in[5], (const float*)d_in[7],
      (const float*)d_in[9], wPin);
  pack3_k<<<(SS * G3 * HH + 255) / 256, 256, 0, stream>>>(
      (const float*)d_in[13], (const float*)d_in[15], (const float*)d_in[17],
      wPhid);
  cvt(d_in[11], wWig,  SS * SS * DD);
  cvt(d_in[19], wWhg,  SS * SS * CATW);
  cvt(d_in[21], wWhc,  SS * CATW * CATW);
  cvt(d_in[23], wWlst, OUTN * FEATW);

  auto gemm = [&](const u16* A, int lda, const u16* W, int ldw, int N,
                  float* C, int ldc, int K, int accum) {
    dim3 grid((N + 63) / 64, BB / 128);
    gemm_bf16<<<grid, 256, 0, stream>>>(A, lda, W, ldw, N, C, ldc, K, accum);
  };

  const float* bi_f = (const float*)d_in[4];
  const float* bi_i = (const float*)d_in[6];
  const float* bi_o = (const float*)d_in[8];
  const float* bi_c = (const float*)d_in[10];
  const float* bi_g = (const float*)d_in[12];
  const float* bh_f = (const float*)d_in[14];
  const float* bh_i = (const float*)d_in[16];
  const float* bh_o = (const float*)d_in[18];
  const float* bh_g = (const float*)d_in[20];
  const float* bh_c = (const float*)d_in[22];

  for (int t = 0; t < REPT; ++t) {
    build_hcat_k<<<(SS * BB * HH + 255) / 256, 256, 0, stream>>>(hbuf, hcat);
    for (int l = 0; l < SS; ++l) {
      const u16* in_h = (l == 0) ? xb : hbuf + (size_t)(l - 1) * BB * HH;
      const u16* hp   = hbuf + (size_t)l * BB * HH;
      // Packed gate pre-activations: P4 = in_h @ [Wif;Wii;Wio;Wic]^T  (N=1024)
      gemm(in_h, DD, wPin + (size_t)l * G4 * DD, DD, G4, P4, G4, DD, 0);
      // Accumulate hidden projections into f/i/o columns  (N=768, ldc=1024)
      gemm(hp, HH, wPhid + (size_t)l * G3 * HH, HH, G3, P4, G4, HH, 1);
      // Big cross-layer matmul: aux3 = hcat @ Wh_c[l]^T   [B,768]x[768,768]
      gemm(hcat, CATW, wWhc + (size_t)l * CATW * CATW, CATW, CATW,
           aux3, CATW, CATW, 0);
      // tiny N=3 gate
      g_gate_k<<<(BB * SS + 255) / 256, 256, 0, stream>>>(
          in_h, hcat, wWig + (size_t)l * SS * DD, bi_g + l * SS,
          wWhg + (size_t)l * SS * CATW, bh_g + l * SS, gbuf);
      u16* featp = (l == SS - 1 && t < REPT - 1) ? feat + (size_t)t * HH : nullptr;
      combine_k<<<(BB * HH + 255) / 256, 256, 0, stream>>>(
          P4, aux3, gbuf,
          bi_f + l * HH, bh_f + l * HH, bi_i + l * HH, bh_i + l * HH,
          bi_o + l * HH, bh_o + l * HH, bi_c + l * HH,
          bh_c + (size_t)l * CATW,
          cbuf + (size_t)l * BB * HH, hbuf + (size_t)l * BB * HH, featp);
    }
  }

  // Output head: [B,2304] x [44,2304]^T, then bias+sigmoid
  gemm(feat, FEATW, wWlst, FEATW, OUTN, lastP, OUTN, FEATW, 0);
  final_out_k<<<(BB * OUTN + 255) / 256, 256, 0, stream>>>(
      lastP, (const float*)d_in[24], (float*)d_out);
}
